// StrictDualCausalCrossAttention_4243427688652
// MI455X (gfx1250) — matrix operand order repaired and checked by reading, not compiled, not run on hardware
//
#include <hip/hip_runtime.h>
#include <hip/hip_bf16.h>

// ---------------------------------------------------------------------------
// StrictDualCausalCrossAttention for MI455X (gfx1250), wave32 + WMMA bf16.
//
//   1) proj_x_f32_to_bf16 : Q=(qb@Wq^T+bq), K=(kvb@Wk^T+bk)  -> (B,S,E) bf16
//                           V=(kvb@Wv^T+bv) stored TRANSPOSED -> (B,E,SKVP)
//   2) attn_kernel        : per (b, head, 16-row q tile), one wave32.
//                           K tiles streamed into LDS with async-to-LDS
//                           (double buffered, s_wait_asynccnt), two-pass
//                           softmax, PV via WMMA, head-mean via atomicAdd
//                           into an accumulator that lives in the 192MB L2.
//   3) proj_o_bf16_to_f32 : out = O@Wo^T+bo stored as (SQ,B,E) fp32.
//
// WS (bf16): Qb[B*SQ*E] | Kb[B*SKV*E] | Vt[B*E*SKVP] | Ob[B*SQ*E]  (~67.4MB)
// ---------------------------------------------------------------------------

typedef __bf16 bf16;
typedef __attribute__((ext_vector_type(16))) __bf16 v16bf;
typedef __attribute__((ext_vector_type(8)))  float  v8f;

#define NHEADS 16
#define SQc    2048
#define Bc     4
#define Ec     1024
#define SKVc   2049
#define SKVP   2080         // padded kv pitch for V^T rows (>= 2048+32)
#define Dc     64
#define QTILES (SQc / 16)   // 128
#define KP     88           // LDS K-tile row pitch (bf16): 16B-aligned rows,
                            // 44-word stride -> 16 distinct banks per half-wave

#if defined(__HIP_DEVICE_COMPILE__) && \
    __has_builtin(__builtin_amdgcn_global_load_async_to_lds_b128)
#define USE_ASYNC_LDS 1
#else
#define USE_ASYNC_LDS 0
#endif

__device__ __forceinline__ v8f wmma_bf16(v16bf a, v16bf b, v8f c) {
  return __builtin_amdgcn_wmma_f32_16x16x32_bf16(
      false, a, false, b, (short)0, c, false, false);
}

#if USE_ASYNC_LDS
// clang prototype (from diagnostics): first arg is v4i* in AS1 ("__device__"),
// second is the LDS destination; offset/cpol are immediates.
typedef int v4i __attribute__((vector_size(16)));
typedef __attribute__((address_space(1))) v4i* gv4i_p;
typedef __attribute__((address_space(3))) v4i* lv4i_p;

__device__ __forceinline__ void wait_async_le4() {
#if __has_builtin(__builtin_amdgcn_s_wait_asynccnt)
  __builtin_amdgcn_s_wait_asynccnt(4);
#else
  asm volatile("s_wait_asynccnt 0x4" ::: "memory");
#endif
}
__device__ __forceinline__ void wait_async_le0() {
#if __has_builtin(__builtin_amdgcn_s_wait_asynccnt)
  __builtin_amdgcn_s_wait_asynccnt(0);
#else
  asm volatile("s_wait_asynccnt 0x0" ::: "memory");
#endif
}
#endif

// ---------------------------------------------------------------------------
// Kernel 1: Y = X(S,B,E) @ W^T + bias (fp32 in, bf16 out).
// transposed==0: Y[(b*S+s)*E + n]          (B,S,E)
// transposed==1: Y[(b*E+n)*SKVP + s]       (B,E,SKVP)  -- for V^T
// Block = 128 threads = 4 waves; each wave owns a 16-col N tile.
// ---------------------------------------------------------------------------
__global__ __launch_bounds__(128) void proj_x_f32_to_bf16(
    const float* __restrict__ X, const float* __restrict__ W,
    const float* __restrict__ bias, bf16* __restrict__ Y, int S,
    int transposed)
{
  const int M    = Bc * S;
  const int lane = threadIdx.x & 31;
  const int wave = threadIdx.x >> 5;
  const int half = lane >> 4;
  const int lm   = lane & 15;
  const int mt   = blockIdx.x;
  const int n0   = (blockIdx.y * 4 + wave) * 16;

  int m = mt * 16 + lm;
  if (m >= M) m = M - 1;                     // clamp loads; stores guarded
  const int b = m / S, s = m - b * S;
  const float* xrow = X + ((size_t)s * Bc + b) * Ec;   // (S,B,E) gather
  const float* wrow = W + (size_t)(n0 + lm) * Ec;      // B col n = W row n

  v8f acc = {};
#pragma unroll 4
  for (int k0 = 0; k0 < Ec; k0 += 32) {
    const float* ap = xrow + k0 + half * 16; // A: element i -> k = half*16+i
    const float* bp = wrow + k0 + half * 16; // B: element i -> k = half*16+i
    __builtin_prefetch(bp + 256, 0, 2);      // global_prefetch_b8
    v16bf a, bb;
#pragma unroll
    for (int i = 0; i < 8; ++i) { a[i] = (bf16)ap[i]; a[8 + i] = (bf16)ap[8 + i]; }
#pragma unroll
    for (int i = 0; i < 16; ++i) bb[i] = (bf16)bp[i];
    acc = wmma_bf16(a, bb, acc);
  }

  const float bv = bias[n0 + lm];
#pragma unroll
  for (int v = 0; v < 8; ++v) {              // C: row = half*8+v, col = lm
    const int row = mt * 16 + half * 8 + v;
    if (row < M) {
      const bf16 val = (bf16)(acc[v] + bv);
      if (!transposed) {
        Y[(size_t)row * Ec + n0 + lm] = val;              // row = b*S+s
      } else {
        const int rb = row / S, rs = row - rb * S;
        Y[((size_t)rb * Ec + n0 + lm) * SKVP + rs] = val; // V^T
      }
    }
  }
}

// ---------------------------------------------------------------------------
// Kernel 2: attention. One wave32 per (b, head, 16-row q tile).
// ---------------------------------------------------------------------------
__global__ __launch_bounds__(32) void attn_kernel(
    const bf16* __restrict__ Qb, const bf16* __restrict__ Kb,
    const bf16* __restrict__ Vt, const unsigned char* __restrict__ cond,
    float* __restrict__ attn_mean /* (B,SQ,SKV) pre-zeroed */,
    bf16* __restrict__ Ob /* (B,SQ,E) */)
{
  __shared__ bf16 pshm[16 * 32];             // P tile: C-layout -> A-layout
#if USE_ASYNC_LDS
  __shared__ __align__(16) bf16 ktile[2][16 * KP];  // double-buffered K tiles
#endif

  const int blk  = blockIdx.x;
  const int qt   = blk % QTILES;
  const int t    = blk / QTILES;
  const int h    = t % NHEADS;
  const int b    = t / NHEADS;
  const int lane = threadIdx.x & 31;
  const int half = lane >> 4;
  const int lm   = lane & 15;

  const bf16* Qbh = Qb + (size_t)b * SQc  * Ec + (size_t)h * Dc;
  const bf16* Kbh = Kb + (size_t)b * SKVc * Ec + (size_t)h * Dc;
  const bf16* Vth = Vt + ((size_t)b * Ec + (size_t)h * Dc) * SKVP;
  const unsigned char* condb = cond + (size_t)b * SQc;

  // Q A-fragments: row m = lm, d = 0..63 split into two K=32 fragments
  const bf16* qrow = Qbh + (size_t)(qt * 16 + lm) * Ec;
  v16bf aq0, aq1;
#pragma unroll
  for (int i = 0; i < 16; ++i) {
    const int k = (half << 4) + i;
    aq0[i] = qrow[k];
    aq1[i] = qrow[32 + k];
  }

  bool ci[8];
#pragma unroll
  for (int v = 0; v < 8; ++v) ci[v] = condb[qt * 16 + half * 8 + v] != 0;

  const float scale = 0.125f;                // 1/sqrt(64)
  const float NEG   = -__builtin_inff();

#if USE_ASYNC_LDS
  // stream one 16(kv) x 64(d) K tile into LDS: 4 async b128 ops per wave
  auto issue_ktile = [&](int buf, int kv0) {
#pragma unroll
    for (int tt = 0; tt < 4; ++tt) {
      const int c    = tt * 32 + lane;       // 0..127 chunk id
      const int row  = c >> 3;               // kv row in tile
      const int coff = c & 7;                // 16B chunk within 128B row
      int kvr = kv0 + row;
      if (kvr >= SKVc) kvr = SKVc - 1;
      const bf16* src = Kbh + (size_t)kvr * Ec + coff * 8;
      __builtin_amdgcn_global_load_async_to_lds_b128(
          (gv4i_p)src, (lv4i_p)&ktile[buf][row * KP + coff * 8], 0, 0);
    }
  };
#endif

  // raw 16x16 score tile: lane col = kv (tile row), rows via WMMA C layout
  auto raw_scores = [&](int buf, int kv0) -> v8f {
    v16bf b0, b1;
#if USE_ASYNC_LDS
    (void)kv0;
    const bf16* kr = &ktile[buf][lm * KP + half * 16];
#pragma unroll
    for (int i = 0; i < 16; ++i) { b0[i] = kr[i]; b1[i] = kr[32 + i]; }
#else
    (void)buf;
    int kvc = kv0 + lm;
    if (kvc >= SKVc) kvc = SKVc - 1;
    const bf16* kr = Kbh + (size_t)kvc * Ec + half * 16;
#pragma unroll
    for (int i = 0; i < 16; ++i) { b0[i] = kr[i]; b1[i] = kr[32 + i]; }
#endif
    v8f s = {};
    s = wmma_bf16(aq0, b0, s);
    s = wmma_bf16(aq1, b1, s);
    return s;
  };

  auto apply_mask = [&](v8f& s, int kv0) {
    const int  kvc    = kv0 + lm;
    const bool valid  = kvc < SKVc;
    const bool iszero = (kvc == 0);
    bool cj = false;
    if (valid && !iszero) cj = condb[kvc - 1] != 0;
#pragma unroll
    for (int v = 0; v < 8; ++v) {
      const int qv = qt * 16 + half * 8 + v;
      bool masked;
      if (!valid)      masked = true;
      else if (iszero) masked = false;       // text column always visible
      else {
        const int  j      = kvc - 1;
        const bool causal = qv < j;
        masked = (causal && !(ci[v] && cj)) || (ci[v] && !cj);
      }
      s[v] = masked ? NEG : s[v] * scale;
    }
  };

  // ---- pass 1: streaming row max / row sum ----
  float m_i[8], l_i[8];
#pragma unroll
  for (int v = 0; v < 8; ++v) { m_i[v] = NEG; l_i[v] = 0.f; }

  int cur = 0;
#if USE_ASYNC_LDS
  issue_ktile(0, 0);
#endif
#pragma unroll 1
  for (int kv0 = 0; kv0 < SKVc; kv0 += 16) {
#if USE_ASYNC_LDS
    const bool more = (kv0 + 16) < SKVc;
    if (more) { issue_ktile(cur ^ 1, kv0 + 16); wait_async_le4(); }
    else      { wait_async_le0(); }
#endif
    v8f s = raw_scores(cur, kv0);
    apply_mask(s, kv0);
#pragma unroll
    for (int v = 0; v < 8; ++v) {
      float x = s[v];                        // reduce within 16-lane half
      x = fmaxf(x, __shfl_xor(x, 1));
      x = fmaxf(x, __shfl_xor(x, 2));
      x = fmaxf(x, __shfl_xor(x, 4));
      x = fmaxf(x, __shfl_xor(x, 8));
      const float mn = fmaxf(m_i[v], x);
      float e = __expf(s[v] - mn);           // -inf -> 0
      e += __shfl_xor(e, 1); e += __shfl_xor(e, 2);
      e += __shfl_xor(e, 4); e += __shfl_xor(e, 8);
      l_i[v] = l_i[v] * __expf(m_i[v] - mn) + e;
      m_i[v] = mn;
    }
    cur ^= 1;
  }

  float linv[8];
#pragma unroll
  for (int v = 0; v < 8; ++v) linv[v] = 1.0f / l_i[v];

  // ---- pass 2: exact probabilities -> head-mean atomics + PV WMMA ----
  v8f o[4] = {v8f{}, v8f{}, v8f{}, v8f{}};

  // PV over one 32-wide kv chunk staged in pshm
  auto pv_accum = [&](int c0) {
    v16bf pa;
#pragma unroll
    for (int i = 0; i < 16; ++i) {
      const int kl = (half << 4) + i;
      pa[i] = pshm[lm * 32 + kl];
    }
#pragma unroll
    for (int dt = 0; dt < 4; ++dt) {         // B[kl][n] = V^T[d = dt*16+n][kv]
      v16bf bv;
      const bf16* vr = Vth + (size_t)(dt * 16 + lm) * SKVP + c0 + half * 16;
#pragma unroll
      for (int i = 0; i < 16; ++i) bv[i] = vr[i];   // contiguous (V^T!)
      o[dt] = wmma_bf16(pa, bv, o[dt]);
    }
  };

  cur = 0;
#if USE_ASYNC_LDS
  issue_ktile(0, 0);
#endif
#pragma unroll 1
  for (int kv0 = 0, ti = 0; kv0 < SKVc; kv0 += 16, ++ti) {
#if USE_ASYNC_LDS
    const bool more = (kv0 + 16) < SKVc;
    if (more) { issue_ktile(cur ^ 1, kv0 + 16); wait_async_le4(); }
    else      { wait_async_le0(); }
#endif
    v8f s = raw_scores(cur, kv0);
    apply_mask(s, kv0);

    const int kvc  = kv0 + lm;
    const int tpar = ti & 1;
#pragma unroll
    for (int v = 0; v < 8; ++v) {
      const float p  = __expf(s[v] - m_i[v]) * linv[v];  // masked/OOB -> 0
      const int   qv = qt * 16 + half * 8 + v;
      if (kvc < SKVc)
        atomicAdd(&attn_mean[((size_t)b * SQc + qv) * SKVc + kvc],
                  p * (1.0f / NHEADS));
      pshm[(half * 8 + v) * 32 + tpar * 16 + lm] = (bf16)p;
    }
    if (tpar == 1) {
      __syncthreads();
      pv_accum(kv0 - 16);
      __syncthreads();
    }
    cur ^= 1;
  }
  // flush: 129 tiles -> last (partial) chunk base 2048, cols 16..31 zeroed
  {
#pragma unroll
    for (int v = 0; v < 8; ++v)
      pshm[(half * 8 + v) * 32 + 16 + lm] = (bf16)0.f;
    __syncthreads();
    pv_accum((SKVc / 16) * 16);              // 2048
  }

  // store normalized O tile
  bf16* obase = Ob + (size_t)b * SQc * Ec + (size_t)h * Dc;
#pragma unroll
  for (int dt = 0; dt < 4; ++dt)
#pragma unroll
    for (int v = 0; v < 8; ++v) {
      const int qv = qt * 16 + half * 8 + v;
      obase[(size_t)qv * Ec + dt * 16 + lm] = (bf16)o[dt][v];
    }
}

// ---------------------------------------------------------------------------
// Kernel 3: out(SQ,B,E) = O(B,SQ,E) @ Wo^T + bo   (bf16 in, fp32 out)
// ---------------------------------------------------------------------------
__global__ __launch_bounds__(128) void proj_o_bf16_to_f32(
    const bf16* __restrict__ Xb, const float* __restrict__ W,
    const float* __restrict__ bias, float* __restrict__ Y)
{
  const int lane = threadIdx.x & 31;
  const int wave = threadIdx.x >> 5;
  const int half = lane >> 4;
  const int lm   = lane & 15;
  const int mt   = blockIdx.x;
  const int n0   = (blockIdx.y * 4 + wave) * 16;

  const int m = mt * 16 + lm;                // M = B*SQ, multiple of 16
  const bf16*  xrow = Xb + (size_t)m * Ec;
  const float* wrow = W + (size_t)(n0 + lm) * Ec;

  v8f acc = {};
#pragma unroll 4
  for (int k0 = 0; k0 < Ec; k0 += 32) {
    const bf16*  ap = xrow + k0 + half * 16;
    const float* bp = wrow + k0 + half * 16;
    __builtin_prefetch(bp + 256, 0, 2);
    v16bf a, bb;
#pragma unroll
    for (int i = 0; i < 8; ++i) { a[i] = ap[i]; a[8 + i] = ap[8 + i]; }
#pragma unroll
    for (int i = 0; i < 16; ++i) bb[i] = (bf16)bp[i];
    acc = wmma_bf16(a, bb, acc);
  }

  const float bv = bias[n0 + lm];
#pragma unroll
  for (int v = 0; v < 8; ++v) {
    const int row = mt * 16 + half * 8 + v;  // row = b*SQ + s
    const int rb = row / SQc, rs = row - rb * SQc;
    Y[((size_t)rs * Bc + rb) * Ec + n0 + lm] = acc[v] + bv;  // (SQ,B,E)
  }
}

// ---------------------------------------------------------------------------
extern "C" void kernel_launch(void* const* d_in, const int* in_sizes, int n_in,
                              void* d_out, int out_size, void* d_ws, size_t ws_size,
                              hipStream_t stream) {
  (void)in_sizes; (void)n_in; (void)out_size; (void)ws_size;

  const float* q    = (const float*)d_in[0];
  const float* kv   = (const float*)d_in[1];
  const unsigned char* cond = (const unsigned char*)d_in[2];   // bool mask
  const float* Wq = (const float*)d_in[3];
  const float* bq = (const float*)d_in[4];
  const float* Wk = (const float*)d_in[5];
  const float* bk = (const float*)d_in[6];
  const float* Wv = (const float*)d_in[7];
  const float* bv = (const float*)d_in[8];
  const float* Wo = (const float*)d_in[9];
  const float* bo = (const float*)d_in[10];

  float* out  = (float*)d_out;                        // (SQ,B,E)
  float* attn = out + (size_t)SQc * Bc * Ec;          // (B,SQ,SKV)

  bf16* Qb = (bf16*)d_ws;
  bf16* Kb = Qb + (size_t)Bc * SQc  * Ec;
  bf16* Vt = Kb + (size_t)Bc * SKVc * Ec;             // (B,E,SKVP)
  bf16* Ob = Vt + (size_t)Bc * Ec * SKVP;

  (void)hipMemsetAsync(attn, 0, (size_t)Bc * SQc * SKVc * sizeof(float), stream);

  const dim3 pblk(128);
  const int mtq  = (Bc * SQc  + 15) / 16;   // 512
  const int mtkv = (Bc * SKVc + 15) / 16;   // 513

  proj_x_f32_to_bf16<<<dim3(mtq,  Ec / 64), pblk, 0, stream>>>(q,  Wq, bq, Qb, SQc,  0);
  proj_x_f32_to_bf16<<<dim3(mtkv, Ec / 64), pblk, 0, stream>>>(kv, Wk, bk, Kb, SKVc, 0);
  proj_x_f32_to_bf16<<<dim3(mtkv, Ec / 64), pblk, 0, stream>>>(kv, Wv, bv, Vt, SKVc, 1);

  attn_kernel<<<dim3(Bc * NHEADS * QTILES), dim3(32), 0, stream>>>(
      Qb, Kb, Vt, cond, attn, Ob);

  proj_o_bf16_to_f32<<<dim3(mtq, Ec / 64), pblk, 0, stream>>>(Ob, Wo, bo, out);
}
